// TransformerSTUHybridBlock_4191888081028
// MI455X (gfx1250) — matrix operand order repaired and checked by reading, not compiled, not run on hardware
//
#include <hip/hip_runtime.h>
#include <hip/hip_bf16.h>

// ---------------------------------------------------------------------------
// TransformerSTUHybridBlock for MI455X (gfx1250, wave32, WMMA)
// B=4 L=2048 D=1024 H=16 HD=64 K=24 MLP=4096
// ---------------------------------------------------------------------------

typedef __bf16 bf16;
typedef __attribute__((ext_vector_type(8)))  __bf16 bf16x8;
typedef __attribute__((ext_vector_type(16))) __bf16 bf16x16;
typedef __attribute__((ext_vector_type(8)))  float  f32x8;

// A-fragment (16x32 bf16): lane = (m = lane%16, half = lane/16).
// VGPRs 0..3 hold K = half*8 + 0..7 ; VGPRs 4..7 hold K = 16 + half*8 + 0..7.
__device__ __forceinline__ bf16x16 frag_a(const bf16* p, int half) {
  bf16x8 lo = *(const bf16x8*)(p + half * 8);
  bf16x8 hi = *(const bf16x8*)(p + 16 + half * 8);
  bf16x16 r;
#pragma unroll
  for (int i = 0; i < 8; ++i) { r[i] = lo[i]; r[i + 8] = hi[i]; }
  return r;
}

// B-fragment (32x16 bf16) from B^T stored row-major [N][K]:
// lane = (n = lane%16, half = lane/16) holds K = half*8 + 0..7 and 16 + half*8 + 0..7.
__device__ __forceinline__ bf16x16 frag_b(const bf16* p, int half) {
  bf16x8 lo = *(const bf16x8*)(p + half * 8);
  bf16x8 hi = *(const bf16x8*)(p + 16 + half * 8);
  bf16x16 r;
#pragma unroll
  for (int i = 0; i < 8; ++i) { r[i] = lo[i]; r[i + 8] = hi[i]; }
  return r;
}

__device__ __forceinline__ float gelu_tanh(float x) {
  float x3 = x * x * x;
  return 0.5f * x * (1.0f + tanhf(0.7978845608028654f * (x + 0.044715f * x3)));
}

// --------------------------- utility kernels -------------------------------

__global__ void k_cvt_bf16(const float* __restrict__ s, bf16* __restrict__ d, int n) {
  int i = blockIdx.x * blockDim.x + threadIdx.x;
  if (i < n) d[i] = (bf16)s[i];
}

// dst[n*K + k] = (bf16)src[k*N + n]   (weight [K,N] f32 -> B^T [N,K] bf16)
__global__ void k_transpose_bf16(const float* __restrict__ src, bf16* __restrict__ dst,
                                 int K, int N) {
  int i = blockIdx.x * blockDim.x + threadIdx.x;
  if (i >= K * N) return;
  int n = i % N, k = i / N;
  dst[(size_t)n * K + k] = (bf16)src[i];
}

// filters[l,d] = sum_k eig_vecs[l,k] * eig_vals[k]^0.25 * w_filters[k,d]
__global__ void k_filters(const float* __restrict__ ev, const float* __restrict__ evec,
                          const float* __restrict__ wf, float* __restrict__ out,
                          int L, int D, int K) {
  int i = blockIdx.x * blockDim.x + threadIdx.x;
  if (i >= L * D) return;
  int d = i % D, l = i / D;
  float acc = 0.f;
  for (int k = 0; k < K; ++k)
    acc += evec[l * K + k] * powf(ev[k], 0.25f) * wf[k * D + d];
  out[i] = acc;
}

// causal per-channel conv: out[b,l,d] = sum_{s<=l} filters[l-s,d] * xi[b,s,d]
// 4 outputs (consecutive l) per thread: each xi[s,d] load feeds 4 FMAs.
__global__ void k_conv(const float* __restrict__ filt, const float* __restrict__ xi,
                       float* __restrict__ out, int L, int D) {
  int i = blockIdx.x * blockDim.x + threadIdx.x;
  int d  = i % D;
  int lq = (i / D) % (L / 4);
  int b  = i / (D * (L / 4));
  const int lbase = lq * 4;

  const float* xp = xi + (size_t)b * L * D + d;        // xi[s,d], ascending s
  const float* fp = filt + (size_t)lbase * D + d;      // filt[lbase - s, d], descending

  float acc0 = 0.f, acc1 = 0.f, acc2 = 0.f, acc3 = 0.f;
#pragma unroll 2
  for (int s = 0; s <= lbase; ++s) {
    __builtin_prefetch(xp + 16 * D, 0, 0);             // global_prefetch_b8
    __builtin_prefetch(fp - 16 * D, 0, 0);
    float x = xp[0];
    acc0 += fp[0]             * x;                     // filt[lbase   - s]
    acc1 += fp[(size_t)D]     * x;                     // filt[lbase+1 - s]
    acc2 += fp[(size_t)2 * D] * x;                     // filt[lbase+2 - s]
    acc3 += fp[(size_t)3 * D] * x;                     // filt[lbase+3 - s]
    xp += D; fp -= D;
  }
  // tail: s = lbase+1 .. lbase+3 (xp now at s = lbase+1)
  {
    float f0 = filt[d], f1 = filt[(size_t)D + d], f2 = filt[(size_t)2 * D + d];
    float x1 = xp[0], x2 = xp[(size_t)D], x3 = xp[(size_t)2 * D];
    acc1 += f0 * x1;
    acc2 += f1 * x1 + f0 * x2;
    acc3 += f2 * x1 + f1 * x2 + f0 * x3;
  }
  size_t o = ((size_t)b * L + lbase) * D + d;
  out[o]                 = acc0;
  out[o + (size_t)D]     = acc1;
  out[o + 2 * (size_t)D] = acc2;
  out[o + 3 * (size_t)D] = acc3;
}

// layernorm row of length D, bf16 output
__global__ __launch_bounds__(256)
void k_ln(const float* __restrict__ x, const float* __restrict__ sc,
          const float* __restrict__ bi, bf16* __restrict__ out, int D) {
  int row = blockIdx.x;
  int t = threadIdx.x;
  const float* xr = x + (size_t)row * D;
  float s = 0.f, s2 = 0.f;
  for (int d = t; d < D; d += 256) { float v = xr[d]; s += v; s2 += v * v; }
  __shared__ float rs[256], rq[256];
  rs[t] = s; rq[t] = s2;
  __syncthreads();
  for (int off = 128; off > 0; off >>= 1) {
    if (t < off) { rs[t] += rs[t + off]; rq[t] += rq[t + off]; }
    __syncthreads();
  }
  float mean = rs[0] / (float)D;
  float var  = rq[0] / (float)D - mean * mean;
  float inv  = rsqrtf(var + 1e-6f);
  for (int d = t; d < D; d += 256)
    out[(size_t)row * D + d] = (bf16)((xr[d] - mean) * inv * sc[d] + bi[d]);
}

// ------------------------------- GEMM --------------------------------------
// C[M,N] = act(scale * A[M,K] @ B[K,N] + bias) + res ; A bf16 [M][K], BT bf16 [N][K]
// 128 threads = 4 waves; block tile 64x128; wave tile 32x64 (2x4 WMMA frags).
// K is a template constant so all fragment-row offsets are load immediates:
// only two loop-carried pointers -> scalar-add addressing, no WMMA/VALU hazards.
template <int K>
__global__ __launch_bounds__(128)
void k_gemm(const bf16* __restrict__ A, const bf16* __restrict__ BT,
            float* __restrict__ Cf, bf16* __restrict__ Cb,
            const float* __restrict__ bias, const float* __restrict__ res,
            int M, int N, float scale, int gelu) {
  const int lane = threadIdx.x & 31;
  const int w    = threadIdx.x >> 5;
  const int ln   = lane & 15;
  const int half = lane >> 4;
  const int m0 = blockIdx.y * 64 + (w >> 1) * 32;
  const int n0 = blockIdx.x * 128 + (w & 1) * 64;

  f32x8 acc[2][4] = {};

  const bf16* ap = A  + (size_t)(m0 + ln) * K;   // frag rows at +0, +16*K
  const bf16* bp = BT + (size_t)(n0 + ln) * K;   // frag rows at +0,16K,32K,48K

#pragma unroll 2
  for (int kk = 0; kk < K; kk += 32) {
    bf16x16 af[2];
    bf16x16 bg[4];
#pragma unroll
    for (int mf = 0; mf < 2; ++mf) af[mf] = frag_a(ap + mf * 16 * K + kk, half);
#pragma unroll
    for (int nf = 0; nf < 4; ++nf) bg[nf] = frag_b(bp + nf * 16 * K + kk, half);
#pragma unroll
    for (int nf = 0; nf < 4; ++nf) {
      acc[0][nf] = __builtin_amdgcn_wmma_f32_16x16x32_bf16(false, af[0], false, bg[nf], (short)0, acc[0][nf], false, false);
      acc[1][nf] = __builtin_amdgcn_wmma_f32_16x16x32_bf16(false, af[1], false, bg[nf], (short)0, acc[1][nf], false, false);
    }
  }

#pragma unroll
  for (int mf = 0; mf < 2; ++mf) {
#pragma unroll
    for (int nf = 0; nf < 4; ++nf) {
#pragma unroll
      for (int j = 0; j < 8; ++j) {
        int row = m0 + mf * 16 + j + 8 * half;   // C layout: VGPR j, half selects M
        int col = n0 + nf * 16 + ln;             // N striped across 16 lanes
        float v = acc[mf][nf][j] * scale;
        if (bias) v += bias[col];
        if (gelu) v = gelu_tanh(v);
        if (res)  v += res[(size_t)row * N + col];
        size_t idx = (size_t)row * N + col;
        if (Cf) Cf[idx] = v; else Cb[idx] = (bf16)v;
      }
    }
  }
}

// --------------------------- flash attention -------------------------------
// q,k,v : bf16 [B, L, H*64]; q pre-scaled by 1/sqrt(HD). Non-causal softmax.
// Block = 128 threads (4 waves), one (b, h, 64-row Q tile); streams 64-row K/V tiles.
__global__ __launch_bounds__(128)
void k_attn(const bf16* __restrict__ q, const bf16* __restrict__ k,
            const bf16* __restrict__ v, bf16* __restrict__ o, int L, int H) {
  alignas(32) __shared__ bf16 ldsP[4 * 16 * 64];   // per-wave P tile (16 x 64)
  alignas(32) __shared__ bf16 ldsVT[64 * 64];      // V tile transposed [hd][kl]

  const int b = blockIdx.z, h = blockIdx.y, q0 = blockIdx.x * 64;
  const int tid = threadIdx.x;
  const int lane = tid & 31, w = tid >> 5;
  const int ln = lane & 15, half = lane >> 4;
  const int DH = H * 64;

  const bf16* qb = q + (size_t)b * L * DH + h * 64;
  const bf16* kb = k + (size_t)b * L * DH + h * 64;
  const bf16* vb = v + (size_t)b * L * DH + h * 64;

  // Q A-fragments for this wave's 16 rows (kept in registers all kernel)
  bf16x16 qa[2];
  {
    const bf16* qr = qb + (size_t)(q0 + w * 16 + ln) * DH;
#pragma unroll
    for (int ks = 0; ks < 2; ++ks) qa[ks] = frag_a(qr + ks * 32, half);
  }

  f32x8 oacc[4] = {};
  float mrow[8], lrow[8];
#pragma unroll
  for (int j = 0; j < 8; ++j) { mrow[j] = -1e30f; lrow[j] = 0.f; }

  const int nkt = L / 64;
  for (int kt = 0; kt < nkt; ++kt) {
    const int k0 = kt * 64;

    // S = Q @ K^T  (K's natural [l][hd] layout is exactly the B^T layout)
    f32x8 s[4];
#pragma unroll
    for (int nf = 0; nf < 4; ++nf) {
      f32x8 a = {};
      const bf16* kr = kb + (size_t)(k0 + nf * 16 + ln) * DH;
#pragma unroll
      for (int ks = 0; ks < 2; ++ks) {
        bf16x16 bg = frag_b(kr + ks * 32, half);
        a = __builtin_amdgcn_wmma_f32_16x16x32_bf16(false, qa[ks], false, bg, (short)0, a, false, false);
      }
      s[nf] = a;
    }

    // online softmax (rows j+8*half; N reduction across 16 lanes of this half)
    float alpha[8];
#pragma unroll
    for (int j = 0; j < 8; ++j) {
      float mx = fmaxf(fmaxf(s[0][j], s[1][j]), fmaxf(s[2][j], s[3][j]));
      mx = fmaxf(mx, __shfl_xor(mx, 1, 32));
      mx = fmaxf(mx, __shfl_xor(mx, 2, 32));
      mx = fmaxf(mx, __shfl_xor(mx, 4, 32));
      mx = fmaxf(mx, __shfl_xor(mx, 8, 32));
      float nm = fmaxf(mrow[j], mx);
      alpha[j] = __expf(mrow[j] - nm);
      mrow[j] = nm;
      float rs = 0.f;
#pragma unroll
      for (int nf = 0; nf < 4; ++nf) {
        float p = __expf(s[nf][j] - nm);
        s[nf][j] = p;
        rs += p;
      }
      rs += __shfl_xor(rs, 1, 32);
      rs += __shfl_xor(rs, 2, 32);
      rs += __shfl_xor(rs, 4, 32);
      rs += __shfl_xor(rs, 8, 32);
      lrow[j] = lrow[j] * alpha[j] + rs;
    }

    __syncthreads();   // previous iteration's LDS consumers are done

    // P -> LDS (C layout -> memory), per-wave region
#pragma unroll
    for (int nf = 0; nf < 4; ++nf)
#pragma unroll
      for (int j = 0; j < 8; ++j)
        ldsP[w * 1024 + (j + 8 * half) * 64 + nf * 16 + ln] = (bf16)s[nf][j];

    // stage V tile transposed: ldsVT[hd][kl]
    for (int c = tid; c < 512; c += 128) {
      int kl = c >> 3, hg = c & 7;
      bf16x8 row = *(const bf16x8*)(vb + (size_t)(k0 + kl) * DH + hg * 8);
#pragma unroll
      for (int e = 0; e < 8; ++e) ldsVT[(hg * 8 + e) * 64 + kl] = row[e];
    }

    __syncthreads();

    // rescale O then O += P @ V
#pragma unroll
    for (int nf = 0; nf < 4; ++nf)
#pragma unroll
      for (int j = 0; j < 8; ++j) oacc[nf][j] *= alpha[j];

#pragma unroll
    for (int pk = 0; pk < 2; ++pk) {
      bf16x16 pa = frag_a(&ldsP[w * 1024 + ln * 64 + pk * 32], half);
#pragma unroll
      for (int nf = 0; nf < 4; ++nf) {
        bf16x16 vg = frag_b(&ldsVT[(nf * 16 + ln) * 64 + pk * 32], half);
        oacc[nf] = __builtin_amdgcn_wmma_f32_16x16x32_bf16(false, pa, false, vg, (short)0, oacc[nf], false, false);
      }
    }
  }

  // normalize and store
#pragma unroll
  for (int j = 0; j < 8; ++j) {
    float inv = 1.0f / lrow[j];
    int row = q0 + w * 16 + j + 8 * half;
#pragma unroll
    for (int nf = 0; nf < 4; ++nf)
      o[(size_t)(b * L + row) * DH + h * 64 + nf * 16 + ln] = (bf16)(oacc[nf][j] * inv);
  }
}

// ------------------------------ launcher -----------------------------------

extern "C" void kernel_launch(void* const* d_in, const int* in_sizes, int n_in,
                              void* d_out, int out_size, void* d_ws, size_t ws_size,
                              hipStream_t stream) {
  (void)in_sizes; (void)n_in; (void)out_size; (void)ws_size;

  const float* inputs    = (const float*)d_in[0];
  const float* eig_vals  = (const float*)d_in[1];
  const float* eig_vecs  = (const float*)d_in[2];
  const float* w_filters = (const float*)d_in[3];
  const float* w_inputs  = (const float*)d_in[4];
  const float* ln1_s     = (const float*)d_in[5];
  const float* ln1_b     = (const float*)d_in[6];
  const float* wq        = (const float*)d_in[7];
  const float* wk        = (const float*)d_in[8];
  const float* wv        = (const float*)d_in[9];
  const float* wo        = (const float*)d_in[10];
  const float* ln2_s     = (const float*)d_in[11];
  const float* ln2_b     = (const float*)d_in[12];
  const float* w1        = (const float*)d_in[13];
  const float* b1        = (const float*)d_in[14];
  const float* w2        = (const float*)d_in[15];
  const float* b2        = (const float*)d_in[16];

  constexpr int B = 4, L = 2048, D = 1024, H = 16, KF = 24, MLP = 4096;
  constexpr int BL = B * L;                     // 8192 rows

  char* ws = (char*)d_ws;
  size_t off = 0;
  auto alloc = [&](size_t bytes) -> char* {
    char* p = ws + off;
    off += (bytes + 255) & ~(size_t)255;
    return p;
  };

  bf16*  inb  = (bf16*)alloc((size_t)BL * D * 2);
  bf16*  wiT  = (bf16*)alloc((size_t)D * D * 2);
  bf16*  wqT  = (bf16*)alloc((size_t)D * D * 2);
  bf16*  wkT  = (bf16*)alloc((size_t)D * D * 2);
  bf16*  wvT  = (bf16*)alloc((size_t)D * D * 2);
  bf16*  woT  = (bf16*)alloc((size_t)D * D * 2);
  bf16*  w1T  = (bf16*)alloc((size_t)D * MLP * 2);
  bf16*  w2T  = (bf16*)alloc((size_t)MLP * D * 2);
  float* filt = (float*)alloc((size_t)L * D * 4);
  float* xi   = (float*)alloc((size_t)BL * D * 4);
  float* stu  = (float*)alloc((size_t)BL * D * 4);
  bf16*  xln  = (bf16*)alloc((size_t)BL * D * 2);
  bf16*  qb   = (bf16*)alloc((size_t)BL * D * 2);
  bf16*  kbuf = (bf16*)alloc((size_t)BL * D * 2);
  bf16*  vbuf = (bf16*)alloc((size_t)BL * D * 2);
  bf16*  ao   = (bf16*)alloc((size_t)BL * D * 2);
  float* xres = (float*)alloc((size_t)BL * D * 4);
  bf16*  yln  = (bf16*)alloc((size_t)BL * D * 2);
  bf16*  h1   = (bf16*)alloc((size_t)BL * MLP * 2);

  // --- prep: conversions / transposes / filters ---
  k_cvt_bf16<<<(BL * D + 255) / 256, 256, 0, stream>>>(inputs, inb, BL * D);
  k_transpose_bf16<<<(D * D + 255) / 256, 256, 0, stream>>>(w_inputs, wiT, D, D);
  k_transpose_bf16<<<(D * D + 255) / 256, 256, 0, stream>>>(wq, wqT, D, D);
  k_transpose_bf16<<<(D * D + 255) / 256, 256, 0, stream>>>(wk, wkT, D, D);
  k_transpose_bf16<<<(D * D + 255) / 256, 256, 0, stream>>>(wv, wvT, D, D);
  k_transpose_bf16<<<(D * D + 255) / 256, 256, 0, stream>>>(wo, woT, D, D);
  k_transpose_bf16<<<(D * MLP + 255) / 256, 256, 0, stream>>>(w1, w1T, D, MLP);
  k_transpose_bf16<<<(MLP * D + 255) / 256, 256, 0, stream>>>(w2, w2T, MLP, D);
  k_filters<<<(L * D + 255) / 256, 256, 0, stream>>>(eig_vals, eig_vecs, w_filters, filt, L, D, KF);

  // --- STU: xi = inputs @ w_inputs ; causal spectral conv -> stu ---
  k_gemm<D><<<dim3(D / 128, BL / 64), 128, 0, stream>>>(inb, wiT, xi, nullptr, nullptr, nullptr,
                                                        BL, D, 1.0f, 0);
  k_conv<<<(B * (L / 4) * D) / 256, 256, 0, stream>>>(filt, xi, stu, L, D);

  // --- attention (pre-LN) ---
  k_ln<<<BL, 256, 0, stream>>>(stu, ln1_s, ln1_b, xln, D);
  k_gemm<D><<<dim3(D / 128, BL / 64), 128, 0, stream>>>(xln, wqT, nullptr, qb, nullptr, nullptr,
                                                        BL, D, 0.125f, 0);   // / sqrt(64)
  k_gemm<D><<<dim3(D / 128, BL / 64), 128, 0, stream>>>(xln, wkT, nullptr, kbuf, nullptr, nullptr,
                                                        BL, D, 1.0f, 0);
  k_gemm<D><<<dim3(D / 128, BL / 64), 128, 0, stream>>>(xln, wvT, nullptr, vbuf, nullptr, nullptr,
                                                        BL, D, 1.0f, 0);
  k_attn<<<dim3(L / 64, H, B), 128, 0, stream>>>(qb, kbuf, vbuf, ao, L, H);
  k_gemm<D><<<dim3(D / 128, BL / 64), 128, 0, stream>>>(ao, woT, xres, nullptr, nullptr, stu,
                                                        BL, D, 1.0f, 0);     // x = a@wo + stu

  // --- MLP (pre-LN) ---
  k_ln<<<BL, 256, 0, stream>>>(xres, ln2_s, ln2_b, yln, D);
  k_gemm<D><<<dim3(MLP / 128, BL / 64), 128, 0, stream>>>(yln, w1T, nullptr, h1, b1, nullptr,
                                                          BL, MLP, 1.0f, 1); // gelu(y@w1+b1)
  k_gemm<MLP><<<dim3(D / 128, BL / 64), 128, 0, stream>>>(h1, w2T, (float*)d_out, nullptr, b2, xres,
                                                          BL, D, 1.0f, 0);   // out = x + (h@w2+b2)
}
